// DEPLoss_old_52793738002794
// MI455X (gfx1250) — matrix operand order repaired and checked by reading, not compiled, not run on hardware
//
#include <hip/hip_runtime.h>

// ---------------------------------------------------------------------------
// DEP loss (HSIC-style) fully fused for gfx1250:
//   dep = (1-e^-1)/(norm*n^2) * sum_c [ S_c - 2 p_c R_c + p_c^2 T ]
// where K_z[i,j] = exp(-0.5*max(0, sq_i+sq_j-2 z_i.z_j)),
//   S_c = sum_{i,j in c} Kz, R_c = sum_{i, j in c} Kz, T = sum R_c.
// Gram computed with v_wmma_f32_16x16x32_bf16 using a hi/lo bf16 split of z.
// Panel staging now uses the Tensor Data Mover (tensor_load_to_lds) with
// hardware LDS padding reproducing the 68-dword anti-conflict stride.
// ---------------------------------------------------------------------------

#define NROWS 8192
#define DDIM  128
#define MT    128            // macrotile edge
#define STR   68             // LDS row stride in dwords (64 data + 4 pad)
#define PANEL (MT * STR)     // 8704 dwords per panel

typedef unsigned int u32;
typedef unsigned long long u64;
typedef __bf16 bf16_t;
typedef bf16_t v16bf __attribute__((ext_vector_type(16)));
typedef float  v8f   __attribute__((ext_vector_type(8)));
typedef u32    v4u   __attribute__((ext_vector_type(4)));
typedef int    v4i   __attribute__((ext_vector_type(4)));
typedef int    v8i   __attribute__((ext_vector_type(8)));

union Frag { v16bf bf; uint4 q[2]; };

__device__ __forceinline__ u32 f2bf_rne(float f) {
    u32 u = __float_as_uint(f);
    return (u + 0x7FFFu + ((u >> 16) & 1u)) >> 16;
}
__device__ __forceinline__ float bf2f(u32 h) { return __uint_as_float(h << 16); }

// ---------------------------------------------------------------------------
// TDM: load a 128-row x 64-dword tile (row stride 64 dwords in memory) into
// LDS with 4 dwords of padding after every 64 dwords (=> 68-dword LDS stride).
// Descriptor built per CDNA5 ISA 8.3-8.6 (D# groups 0..3).
// ---------------------------------------------------------------------------
__device__ __forceinline__ void tdm_load_tile(const u32* gsrc, u32* lds_dst) {
    u64 ga = (u64)(const void*)gsrc;           // 57-bit global byte address
    u32 la = (u32)(u64)(void*)lds_dst;         // LDS byte offset (addr[31:0])

    v4u g0;
    g0.x = 1u;                                  // count=1 (valid), user mode
    g0.y = la;                                  // lds_addr
    g0.z = (u32)ga;                             // global_addr[31:0]
    g0.w = ((u32)(ga >> 32) & 0x1FFFFFFu)       // global_addr[56:32]
         | (2u << 30);                          // type=2 ("image")

    v8i g1;
    g1[0] = (int)((2u << 16)                    // data_size = 4 bytes
                | (1u << 20)                    // pad_enable
                | (5u << 22)                    // pad_interval: 64 dwords
                | (3u << 25));                  // pad_amount: 4 dwords
    g1[1] = (int)(64u << 16);                   // tensor_dim0 = 64 (lo16 @ bits63:48)
    g1[2] = (int)(8192u << 16);                 // tensor_dim1 = 8192 (lo16 @ bits95:80)
    g1[3] = (int)(64u << 16);                   // tile_dim0 = 64 (@ bits127:112)
    g1[4] = (int)128u;                          // tile_dim1 = 128 (@ bits143:128)
    g1[5] = (int)64u;                           // tensor_dim0_stride = 64 (lo32)
    g1[6] = 0;
    g1[7] = 0;

    v4i g2 = { 0, 0, 0, 0 };                    // 2D: dims 2/3 unused
    v4i g3 = { 0, 0, 0, 0 };

#if __clang_major__ >= 23
    v8i gx = { 0, 0, 0, 0, 0, 0, 0, 0 };
    __builtin_amdgcn_tensor_load_to_lds(g0, g1, g2, g3, gx, 0);
#else
    __builtin_amdgcn_tensor_load_to_lds(g0, g1, g2, g3, 0);
#endif
}

// ---------------------------------------------------------------------------
// Kernel 0: zero the 8 global accumulators (R[4], S[4]).
// ---------------------------------------------------------------------------
__global__ void dep_zero_acc(float* __restrict__ gacc) {
    if (threadIdx.x < 8) gacc[threadIdx.x] = 0.0f;
}

// ---------------------------------------------------------------------------
// Kernel 1: per-row preprocess. One wave per row; lane holds one float4.
// Produces packed-bf16 hi/lo arrays and sq[i] = ||hi+lo||^2.
// ---------------------------------------------------------------------------
__global__ void dep_prep(const float* __restrict__ z,
                         u32* __restrict__ hi, u32* __restrict__ lo,
                         float* __restrict__ sq) {
    int wid  = threadIdx.x >> 5;
    int lane = threadIdx.x & 31;
    int row  = blockIdx.x * 8 + wid;

    const float4* z4 = (const float4*)z;
    float4 x = z4[row * 32 + lane];
    float xs[4] = { x.x, x.y, x.z, x.w };

    u32 h[4], l[4];
    float ssum = 0.0f;
#pragma unroll
    for (int i = 0; i < 4; ++i) {
        h[i] = f2bf_rne(xs[i]);
        float hf = bf2f(h[i]);
        float rres = xs[i] - hf;
        l[i] = f2bf_rne(rres);
        float lf = bf2f(l[i]);
        float e = hf + lf;
        ssum += e * e;
    }
    int base = row * 64 + 2 * lane;
    hi[base]     = h[0] | (h[1] << 16);
    hi[base + 1] = h[2] | (h[3] << 16);
    lo[base]     = l[0] | (l[1] << 16);
    lo[base + 1] = l[2] | (l[3] << 16);

#pragma unroll
    for (int o = 16; o; o >>= 1) ssum += __shfl_xor(ssum, o, 32);
    if (lane == 0) sq[row] = ssum;
}

// ---------------------------------------------------------------------------
// Kernel 2: fused Gram + exp + class-sum reduction over lower-triangle
// 128x128 macrotiles. TDM stages 4 bf16 panels; 8 waves each run a 32x64
// register tile = 2x4 accumulators; 96 v_wmma per wave.
// ---------------------------------------------------------------------------
__global__ void __launch_bounds__(256)
dep_main(const u32* __restrict__ hi, const u32* __restrict__ lo,
         const float* __restrict__ sq, const int* __restrict__ s,
         float* __restrict__ gacc) {
    __shared__ __align__(16) u32 sm[4 * PANEL + 1024];

    u32* pAhi = sm;
    u32* pAlo = sm + PANEL;
    u32* pBhi = sm + 2 * PANEL;
    u32* pBlo = sm + 3 * PANEL;
    float* sqA  = (float*)(sm + 4 * PANEL);
    float* sqB  = sqA + 128;
    u32*   clsA = (u32*)(sqB + 128);
    u32*   clsB = clsA + 128;
    float* ohA  = (float*)(clsB + 128);          // [128][4] one-hot of row class

    const int tid = threadIdx.x;
    const int b   = blockIdx.x;

    // lower-triangle block index -> (I, J), I >= J
    int I = (int)((sqrtf(8.0f * (float)b + 1.0f) - 1.0f) * 0.5f);
    while ((I + 1) * (I + 2) / 2 <= b) ++I;
    while (I * (I + 1) / 2 > b) --I;
    const int J = b - I * (I + 1) / 2;
    const int Ibase = I * MT;
    const int Jbase = J * MT;
    const bool offd = (I != J);

    // ---- TDM panel staging: 4 descriptors issued once (wave 0) ----
    if (tid < 32) {
        tdm_load_tile(hi + (u64)Ibase * 64u, pAhi);
        tdm_load_tile(lo + (u64)Ibase * 64u, pAlo);
        tdm_load_tile(hi + (u64)Jbase * 64u, pBhi);
        tdm_load_tile(lo + (u64)Jbase * 64u, pBlo);
    }

    // ---- side-table staging (sq / class / one-hot) ----
    if (tid < 128) {
        sqA[tid] = sq[Ibase + tid];
        u32 cA = (u32)s[Ibase + tid] & 3u;
        clsA[tid] = cA;
        float4 oh = make_float4(cA == 0u ? 1.f : 0.f, cA == 1u ? 1.f : 0.f,
                                cA == 2u ? 1.f : 0.f, cA == 3u ? 1.f : 0.f);
        *(float4*)(ohA + tid * 4) = oh;
    } else {
        int t = tid - 128;
        sqB[t]  = sq[Jbase + t];
        clsB[t] = (u32)s[Jbase + t] & 3u;
    }

    __builtin_amdgcn_s_wait_tensorcnt(0);        // issuing wave drains TDM
    __syncthreads();                             // publish LDS to all waves

    const int wid  = tid >> 5;
    const int lane = tid & 31;
    const int lm   = lane & 15;
    const int half = lane >> 4;
    const int wm   = wid & 3;     // 4 row-groups of 32
    const int wn   = wid >> 2;    // 2 col-groups of 64

    v8f zero8 = { 0.f, 0.f, 0.f, 0.f, 0.f, 0.f, 0.f, 0.f };
    v8f acc[2][4];
#pragma unroll
    for (int mm = 0; mm < 2; ++mm)
#pragma unroll
        for (int nn = 0; nn < 4; ++nn) acc[mm][nn] = zero8;

    // ---- GEMM: K=128 in 4 steps of 32 ----
#pragma unroll
    for (int ks = 0; ks < 4; ++ks) {
        const int tb = ks * 16;                  // dword (bf16-pair) base of K-chunk
        Frag Bh[4], Bl[4];
#pragma unroll
        for (int nn = 0; nn < 4; ++nn) {
            int rr = wn * 64 + nn * 16 + lm;
            const u32* bh = pBhi + rr * STR + tb + half * 8;   // B: t = tb + 8*half + {0..7}
            const u32* bl = pBlo + rr * STR + tb + half * 8;
            Bh[nn].q[0] = *(const uint4*)(bh);
            Bh[nn].q[1] = *(const uint4*)(bh + 4);
            Bl[nn].q[0] = *(const uint4*)(bl);
            Bl[nn].q[1] = *(const uint4*)(bl + 4);
        }
#pragma unroll
        for (int mm = 0; mm < 2; ++mm) {
            int rr = wm * 32 + mm * 16 + lm;
            const u32* ah = pAhi + rr * STR + tb + half * 8;   // A: t = tb + 8*half + {0..7}
            const u32* al = pAlo + rr * STR + tb + half * 8;
            Frag Ah, Al;
            Ah.q[0] = *(const uint4*)(ah);
            Ah.q[1] = *(const uint4*)(ah + 4);
            Al.q[0] = *(const uint4*)(al);
            Al.q[1] = *(const uint4*)(al + 4);
#pragma unroll
            for (int nn = 0; nn < 4; ++nn) {
                acc[mm][nn] = __builtin_amdgcn_wmma_f32_16x16x32_bf16(
                    false, Ah.bf, false, Bh[nn].bf, (short)0, acc[mm][nn], false, false);
                acc[mm][nn] = __builtin_amdgcn_wmma_f32_16x16x32_bf16(
                    false, Ah.bf, false, Bl[nn].bf, (short)0, acc[mm][nn], false, false);
                acc[mm][nn] = __builtin_amdgcn_wmma_f32_16x16x32_bf16(
                    false, Al.bf, false, Bh[nn].bf, (short)0, acc[mm][nn], false, false);
            }
        }
    }

    // ---- epilogue: Kz = exp(-0.5*max(d2,0)); per-class sums ----
    float aR[4] = { 0.f, 0.f, 0.f, 0.f };
    float aS[4] = { 0.f, 0.f, 0.f, 0.f };
    const float wS = offd ? 2.0f : 1.0f;

#pragma unroll
    for (int mm = 0; mm < 2; ++mm) {
#pragma unroll
        for (int nn = 0; nn < 4; ++nn) {
            int jl = wn * 64 + nn * 16 + lm;     // C/D layout: N = lane%16
            float sqj = sqB[jl];
            u32   cj  = clsB[jl];
            float ob0 = (cj == 0u) ? 1.f : 0.f;
            float ob1 = (cj == 1u) ? 1.f : 0.f;
            float ob2 = (cj == 2u) ? 1.f : 0.f;
            float ob3 = (cj == 3u) ? 1.f : 0.f;
#pragma unroll
            for (int r = 0; r < 8; ++r) {
                int il = wm * 32 + mm * 16 + r + half * 8;     // M = r + 8*half
                float d2 = sqA[il] + sqj - 2.0f * acc[mm][nn][r];
                d2 = fmaxf(d2, 0.0f);
                float v = __expf(-0.5f * d2);
                u32 ci = clsA[il];
                float evw = (ci == cj) ? (wS * v) : 0.0f;
                aS[0] += evw * ob0;  aS[1] += evw * ob1;
                aS[2] += evw * ob2;  aS[3] += evw * ob3;
                aR[0] += v * ob0;    aR[1] += v * ob1;
                aR[2] += v * ob2;    aR[3] += v * ob3;
                if (offd) {          // mirrored entry (j,i): column class = ci
                    const float* oa = ohA + il * 4;
                    aR[0] += v * oa[0];  aR[1] += v * oa[1];
                    aR[2] += v * oa[2];  aR[3] += v * oa[3];
                }
            }
        }
    }

    // ---- block reduce + global atomics ----
    __syncthreads();
    float* red = (float*)sm;                     // reuse panel LDS (8 KB)
#pragma unroll
    for (int c = 0; c < 4; ++c) {
        red[tid * 8 + c]     = aR[c];
        red[tid * 8 + 4 + c] = aS[c];
    }
    __syncthreads();
    if (tid < 8) {
        float ssum = 0.0f;
        for (int k = 0; k < 256; ++k) ssum += red[k * 8 + tid];
        atomicAdd(&gacc[tid], ssum);
    }
}

// ---------------------------------------------------------------------------
// Kernel 3: class counts + closed-form combination.
// ---------------------------------------------------------------------------
__global__ void dep_finalize(const float* __restrict__ gacc,
                             const int* __restrict__ s,
                             const float* __restrict__ normp,
                             float* __restrict__ out) {
    __shared__ int cnt[4];
    int tid = threadIdx.x;
    if (tid < 4) cnt[tid] = 0;
    __syncthreads();
    for (int i = tid; i < NROWS; i += 256) atomicAdd(&cnt[s[i] & 3], 1);
    __syncthreads();
    if (tid == 0) {
        float R[4], S[4], T = 0.0f;
#pragma unroll
        for (int c = 0; c < 4; ++c) { R[c] = gacc[c]; S[c] = gacc[4 + c]; T += R[c]; }
        const float n = (float)NROWS;
        float dep = 0.0f;
#pragma unroll
        for (int c = 0; c < 4; ++c) {
            float p = (float)cnt[c] / n;
            dep += S[c] - 2.0f * p * R[c] + p * p * T;
        }
        const float A = 0.63212055882855767f;    // 1 - e^{-1}
        dep = A * dep / (normp[0] * n * n);
        out[0] = dep;
    }
}

// ---------------------------------------------------------------------------
extern "C" void kernel_launch(void* const* d_in, const int* in_sizes, int n_in,
                              void* d_out, int out_size, void* d_ws, size_t ws_size,
                              hipStream_t stream) {
    const float* z    = (const float*)d_in[0];
    const int*   s    = (const int*)d_in[1];
    const float* norm = (const float*)d_in[2];

    char* ws = (char*)d_ws;
    float* gacc = (float*)ws;                               // 8 floats
    float* sq   = (float*)(ws + 4096);                      // 32 KB
    u32*   hi   = (u32*)(ws + 4096 + NROWS * 4);            // 2 MB
    u32*   lo   = (u32*)(ws + 4096 + NROWS * 4 + NROWS * 64 * 4);

    dep_zero_acc<<<1, 32, 0, stream>>>(gacc);
    dep_prep<<<NROWS / 8, 256, 0, stream>>>(z, hi, lo, sq);

    const int nTiles = NROWS / MT;                          // 64
    const int nBlocks = nTiles * (nTiles + 1) / 2;          // 2080 (lower triangle)
    dep_main<<<nBlocks, 256, 0, stream>>>(hi, lo, sq, s, gacc);

    dep_finalize<<<1, 256, 0, stream>>>(gacc, s, norm, (float*)d_out);
}
